// SnLocalDecoder_5059471474987
// MI455X (gfx1250) — matrix operand order repaired and checked by reading, not compiled, not run on hardware
//
#include <hip/hip_runtime.h>

// ---------------------------------------------------------------------------
// SnLocalDecoder on MI455X (gfx1250): bf16 activations + v_wmma_f32_16x16x32_bf16
// B=4, N=48, C=32 hidden, C_OUT=16. All heavy contractions on the WMMA pipe.
// ---------------------------------------------------------------------------

#define NN 48
#define NB2 (NN*NN)        // 2304
#define NB3 (NN*NN*NN)     // 110592
#define BBATCH 4
#define CCH 32
#define RTOT (BBATCH*NB3)  // 442368 rows of the big tensor

typedef __attribute__((ext_vector_type(16))) __bf16        v16bf;
typedef __attribute__((ext_vector_type(8)))  float         v8f;
typedef __attribute__((ext_vector_type(4)))  unsigned int  v4u;
typedef __attribute__((ext_vector_type(2)))  unsigned int  v2u;

union BOp { v4u q[2]; v16bf v; };

__device__ __forceinline__ unsigned short f2bf(float f) {
  union { float f; unsigned u; } a; a.f = f;
  return (unsigned short)((a.u + 0x7FFFu + ((a.u >> 16) & 1u)) >> 16);
}
__device__ __forceinline__ float bf2f(unsigned short h) {
  union { unsigned u; float f; } a; a.u = ((unsigned)h) << 16;
  return a.f;
}
__device__ __forceinline__ v8f wmma_bf16(const BOp& A, const BOp& B, v8f C) {
  return __builtin_amdgcn_wmma_f32_16x16x32_bf16(false, A.v, false, B.v,
                                                 (short)0, C, false, false);
}
__device__ __forceinline__ v8f v8zero() { v8f z = {0.f,0.f,0.f,0.f,0.f,0.f,0.f,0.f}; return z; }

// ---------------------------------------------------------------------------
// valid[b,a] = mask[b,a,a,a]  (mask is AND of per-node validity)
__global__ void k_valid(const unsigned char* __restrict__ mask, float* __restrict__ valid) {
  int t = threadIdx.x;
  if (t < BBATCH * NN) {
    int b = t / NN, a = t % NN;
    long idx = ((long)(b * NN + a) * NN + a) * NN + a;
    valid[t] = mask[idx] ? 1.0f : 0.0f;
  }
}

// ---------------------------------------------------------------------------
// Pre-swizzle a (32 x dout) weight matrix into WMMA B-operand lane layout.
// lane l (g=l/16, n=l%16) holds K = 8g+t (t<8), 16+8g+(t-8) (t>=8), column = 16h + n, t = 0..15.
__global__ void k_swz(const float* __restrict__ M, int dout, int h,
                      unsigned short* __restrict__ dst) {
  int t = blockIdx.x * blockDim.x + threadIdx.x;
  if (t < 512) {
    int lane = t >> 4, p = t & 15;
    int k = ((p >> 3) << 4) + ((lane >> 4) << 3) + (p & 7);
    int d = h * 16 + (lane & 15);
    dst[t] = f2bf(M[k * dout + d]);
  }
}

// ---------------------------------------------------------------------------
// t[b,i,j,m,c] = x[b,i,c]*x[b,j,c]*x[b,m,c] * mask  -> bf16
__global__ void k_build_t(const float* __restrict__ x, const float* __restrict__ valid,
                          unsigned short* __restrict__ tb) {
  long gid = (long)blockIdx.x * blockDim.x + threadIdx.x; // RTOT*8 threads
  long row = gid >> 3;
  int cq = (int)(gid & 7) * 4;
  int m = (int)(row % NN);
  int j = (int)((row / NN) % NN);
  int i = (int)((row / NB2) % NN);
  int b = (int)(row / NB3);
  float fac = valid[b*NN+i] * valid[b*NN+j] * valid[b*NN+m];
  const float* xb = x + (long)b * NN * CCH;
  float p0 = xb[i*CCH+cq+0]*xb[j*CCH+cq+0]*xb[m*CCH+cq+0]*fac;
  float p1 = xb[i*CCH+cq+1]*xb[j*CCH+cq+1]*xb[m*CCH+cq+1]*fac;
  float p2 = xb[i*CCH+cq+2]*xb[j*CCH+cq+2]*xb[m*CCH+cq+2]*fac;
  float p3 = xb[i*CCH+cq+3]*xb[j*CCH+cq+3]*xb[m*CCH+cq+3]*fac;
  v2u o;
  o.x = (unsigned)f2bf(p0) | ((unsigned)f2bf(p1) << 16);
  o.y = (unsigned)f2bf(p2) | ((unsigned)f2bf(p3) << 16);
  *(v2u*)(tb + row * CCH + cq) = o;
}

// ---------------------------------------------------------------------------
// Reductions: rowm[b,a,m,c]=mean_j t, colm[b,a,m,c]=mean_i t, diag=t[b,a,a,m,c]
__global__ void k_red1(const unsigned short* __restrict__ tb,
                       float* __restrict__ rowm, float* __restrict__ colm,
                       float* __restrict__ diagm) {
  int gid = blockIdx.x * blockDim.x + threadIdx.x; // B*N*N*32 = 294912
  int c = gid & 31;
  int m = (gid >> 5) % NN;
  int a = (gid / (32 * NN)) % NN;
  int b = gid / (32 * NB2);
  long bb = (long)b * NB3;
  float rs = 0.f, cs = 0.f;
  for (int q = 0; q < NN; ++q) {
    rs += bf2f(tb[(bb + (long)a * NB2 + (long)q * NN + m) * CCH + c]);
    cs += bf2f(tb[(bb + (long)q * NB2 + (long)a * NN + m) * CCH + c]);
  }
  rowm[gid] = rs * (1.f / 48.f);
  colm[gid] = cs * (1.f / 48.f);
  diagm[gid] = bf2f(tb[(bb + (long)a * NB2 + (long)a * NN + m) * CCH + c]);
}

__global__ void k_tot(const float* __restrict__ rowm, float* __restrict__ totm) {
  int gid = blockIdx.x * blockDim.x + threadIdx.x; // B*N*32 = 6144
  if (gid >= BBATCH * NN * CCH) return;
  int c = gid & 31;
  int m = (gid >> 5) % NN;
  int b = gid / (32 * NN);
  float s = 0.f;
  for (int i = 0; i < NN; ++i) s += rowm[((b * NN + i) * NN + m) * CCH + c];
  totm[gid] = s * (1.f / 48.f);
}

// D1[b,a,m,d] = diag.W2 + rowm.W3 ; D2[b,a,m,d] = colm.W4   (stride 32 in d)
__global__ void k_dmix(const float* __restrict__ diagm, const float* __restrict__ rowm,
                       const float* __restrict__ colm,
                       const float* __restrict__ W2, const float* __restrict__ W3,
                       const float* __restrict__ W4, int dout,
                       float* __restrict__ D1, float* __restrict__ D2) {
  int gid = blockIdx.x * blockDim.x + threadIdx.x;
  if (gid >= BBATCH * NB2 * dout) return;
  int d = gid % dout;
  int m = (gid / dout) % NN;
  int a = (gid / (dout * NN)) % NN;
  int b = gid / (dout * NB2);
  int rb = ((b * NN + a) * NN + m) * 32;
  float s1 = 0.f, s2 = 0.f;
  for (int c = 0; c < CCH; ++c) {
    s1 += diagm[rb + c] * W2[c * dout + d] + rowm[rb + c] * W3[c * dout + d];
    s2 += colm[rb + c] * W4[c * dout + d];
  }
  D1[rb + d] = s1;
  D2[rb + d] = s2;
}

// D3[b,m,d] = tot.W5 + bias   (stride 32 in d)
__global__ void k_d3(const float* __restrict__ totm, const float* __restrict__ W5,
                     const float* __restrict__ bias, int dout, float* __restrict__ D3) {
  int gid = blockIdx.x * blockDim.x + threadIdx.x;
  if (gid >= BBATCH * NN * dout) return;
  int d = gid % dout;
  int m = (gid / dout) % NN;
  int b = gid / (dout * NN);
  float s = bias[d];
  for (int c = 0; c < CCH; ++c) s += totm[(b * NN + m) * 32 + c] * W5[c * dout + d];
  D3[(b * NN + m) * 32 + d] = s;
}

// ---------------------------------------------------------------------------
// Fused: s = mask( t.W0 + tT.W1 + D1 + D2 + D3 ); then via LDS relayout:
// v = s.Wv + bv ; w = s.Ww + bw  (all WMMA). One wave = 16 rows (m) x 32 ch.
__global__ __launch_bounds__(256) void k_snconv_vw(
    const unsigned short* __restrict__ tb, const unsigned short* __restrict__ swz,
    const float* __restrict__ D1, const float* __restrict__ D2, const float* __restrict__ D3,
    const float* __restrict__ valid, const float* __restrict__ bv, const float* __restrict__ bw,
    unsigned short* __restrict__ vout, unsigned short* __restrict__ wout) {
  __shared__ unsigned short sh[8 * 512];
  int tid = threadIdx.x;
  int wv = tid >> 5, lane = tid & 31;
  int g = lane >> 4, n = lane & 15, rl = lane & 15;
  int tile = blockIdx.x * 8 + wv;                 // 27648 tiles total, exact
  int mt = tile % 3;
  int j = (tile / 3) % NN;
  int i = (tile / (3 * NN)) % NN;
  int b = tile / (3 * NB2);
  int m0 = mt * 16;
  long rowA = ((long)(b * NN + i) * NN + j) * NN + m0;
  long rowT = ((long)(b * NN + j) * NN + i) * NN + m0;
  int c0 = 8 * g, c1 = 16 + 8 * g;

  const unsigned short* pA = tb + (rowA + rl) * CCH;
  const unsigned short* pT = tb + (rowT + rl) * CCH;
  BOp A1, A2;
  A1.q[0] = *(const v4u*)(pA + c0); A1.q[1] = *(const v4u*)(pA + c1);
  A2.q[0] = *(const v4u*)(pT + c0); A2.q[1] = *(const v4u*)(pT + c1);

  const unsigned short* sw = swz + lane * 16;     // ops: W0h0,W0h1,W1h0,W1h1,Wv..,Ww..
  BOp W0h[2], W1h[2], Wvh[2], Wwh[2];
#pragma unroll
  for (int h = 0; h < 2; ++h) {
    W0h[h].q[0] = *(const v4u*)(sw + (0 + h) * 512);
    W0h[h].q[1] = *(const v4u*)(sw + (0 + h) * 512 + 8);
    W1h[h].q[0] = *(const v4u*)(sw + (2 + h) * 512);
    W1h[h].q[1] = *(const v4u*)(sw + (2 + h) * 512 + 8);
    Wvh[h].q[0] = *(const v4u*)(sw + (4 + h) * 512);
    Wvh[h].q[1] = *(const v4u*)(sw + (4 + h) * 512 + 8);
    Wwh[h].q[0] = *(const v4u*)(sw + (6 + h) * 512);
    Wwh[h].q[1] = *(const v4u*)(sw + (6 + h) * 512 + 8);
  }

  v8f acc[2];
#pragma unroll
  for (int h = 0; h < 2; ++h) {
    v8f z = v8zero();
    z = wmma_bf16(A1, W0h[h], z);
    z = wmma_bf16(A2, W1h[h], z);
    acc[h] = z;
  }

  float vi = valid[b * NN + i], vj = valid[b * NN + j];
#pragma unroll
  for (int r = 0; r < 8; ++r) {
    int m = m0 + r + 8 * g;
    float fac = vi * vj * valid[b * NN + m];
    int o1 = ((b * NN + i) * NN + m) * 32;
    int o2 = ((b * NN + j) * NN + m) * 32;
    int o3 = (b * NN + m) * 32;
#pragma unroll
    for (int h = 0; h < 2; ++h) {
      int d = n + 16 * h;
      acc[h][r] = (acc[h][r] + D1[o1 + d] + D2[o2 + d] + D3[o3 + d]) * fac;
    }
  }

  // LDS round-trip: C-layout (f32) -> A-operand layout (bf16). Same wave, DS in-order.
  unsigned short* shw = sh + wv * 512;
#pragma unroll
  for (int r = 0; r < 8; ++r) {
    int ml = r + 8 * g;
    shw[ml * 32 + n]      = f2bf(acc[0][r]);
    shw[ml * 32 + n + 16] = f2bf(acc[1][r]);
  }
  BOp SA;
  SA.q[0] = *(const v4u*)(shw + rl * 32 + c0);
  SA.q[1] = *(const v4u*)(shw + rl * 32 + c1);

#pragma unroll
  for (int h = 0; h < 2; ++h) {
    v8f zv = wmma_bf16(SA, Wvh[h], v8zero());
    v8f zw = wmma_bf16(SA, Wwh[h], v8zero());
    float bvv = bv[n + 16 * h], bww = bw[n + 16 * h];
#pragma unroll
    for (int r = 0; r < 8; ++r) {
      long row = rowA + r + 8 * g;
      vout[row * CCH + n + 16 * h] = f2bf(zv[r] + bvv);
      wout[row * CCH + n + 16 * h] = f2bf(zw[r] + bww);
    }
  }
}

// ---------------------------------------------------------------------------
// z[b,i,j,m,c] = (1/48) sum_k v[b,i,k,m,c] * w[b,k,j,m,c]
// One block = (b,m,c); stage V(48x48) and W^T(48x48) in LDS (K zero-padded to 64),
// 9 waves each do one 16x16 tile with two K=32 bf16 WMMAs.
__global__ __launch_bounds__(288) void k_z(const unsigned short* __restrict__ vb,
                                           const unsigned short* __restrict__ wb,
                                           unsigned short* __restrict__ zb) {
  __shared__ unsigned short Vl[48 * 72];
  __shared__ unsigned short Wt[48 * 72];
  int tid = threadIdx.x;
  int cid = blockIdx.x;                 // B*N*32 = 6144
  int c = cid & 31;
  int m = (cid >> 5) % NN;
  int b = cid / (32 * NN);
  for (int e = tid; e < 48 * 72; e += 288) { Vl[e] = 0; Wt[e] = 0; }
  __syncthreads();
  long bb = (long)b * NB3;
  for (int e = tid; e < 48 * 48; e += 288) {
    int a = e / 48, k = e % 48;
    Vl[a * 72 + k] = vb[(bb + (long)a * NB2 + (long)k * NN + m) * CCH + c]; // V[i=a][k]
    Wt[a * 72 + k] = wb[(bb + (long)k * NB2 + (long)a * NN + m) * CCH + c]; // W^T[j=a][k]
  }
  __syncthreads();
  int wv = tid >> 5, lane = tid & 31;
  int g = lane >> 4, n = lane & 15;
  int i0 = (wv / 3) * 16, j0 = (wv % 3) * 16;
  int ri = i0 + (lane & 15);
  int rj = j0 + n;
  v8f acc = v8zero();
#pragma unroll
  for (int k0 = 0; k0 < 64; k0 += 32) {
    BOp Av, Bw;
    Av.q[0] = *(const v4u*)&Vl[ri * 72 + k0 + 8 * g];
    Av.q[1] = *(const v4u*)&Vl[ri * 72 + k0 + 16 + 8 * g];
    Bw.q[0] = *(const v4u*)&Wt[rj * 72 + k0 + 8 * g];
    Bw.q[1] = *(const v4u*)&Wt[rj * 72 + k0 + 16 + 8 * g];
    acc = wmma_bf16(Av, Bw, acc);
  }
#pragma unroll
  for (int r = 0; r < 8; ++r) {
    int ii = i0 + r + 8 * g;
    zb[(bb + (long)ii * NB2 + (long)rj * NN + m) * CCH + c] = f2bf(acc[r] * (1.f / 48.f));
  }
}

// ---------------------------------------------------------------------------
// t_next = mask( relu( z.Wo + bo ) )
__global__ __launch_bounds__(256) void k_wo(const unsigned short* __restrict__ zb,
                                            const unsigned short* __restrict__ swzWo,
                                            const float* __restrict__ bo,
                                            const float* __restrict__ valid,
                                            unsigned short* __restrict__ tout) {
  int tid = threadIdx.x;
  int wv = tid >> 5, lane = tid & 31;
  int g = lane >> 4, n = lane & 15, rl = lane & 15;
  int tile = blockIdx.x * 8 + wv;
  int mt = tile % 3;
  int j = (tile / 3) % NN;
  int i = (tile / (3 * NN)) % NN;
  int b = tile / (3 * NB2);
  int m0 = mt * 16;
  long rowA = ((long)(b * NN + i) * NN + j) * NN + m0;
  int c0 = 8 * g, c1 = 16 + 8 * g;
  const unsigned short* pA = zb + (rowA + rl) * CCH;
  BOp A;
  A.q[0] = *(const v4u*)(pA + c0); A.q[1] = *(const v4u*)(pA + c1);
  const unsigned short* sw = swzWo + lane * 16;
  float vi = valid[b * NN + i], vj = valid[b * NN + j];
#pragma unroll
  for (int h = 0; h < 2; ++h) {
    BOp Bm;
    Bm.q[0] = *(const v4u*)(sw + h * 512);
    Bm.q[1] = *(const v4u*)(sw + h * 512 + 8);
    v8f o = wmma_bf16(A, Bm, v8zero());
    float bb = bo[n + 16 * h];
#pragma unroll
    for (int r = 0; r < 8; ++r) {
      int m = m0 + r + 8 * g;
      float val = o[r] + bb;
      val = val > 0.f ? val : 0.f;
      val *= vi * vj * valid[b * NN + m];
      tout[(rowA + r + 8 * g) * CCH + n + 16 * h] = f2bf(val);
    }
  }
}

// ---------------------------------------------------------------------------
// Final snconv (C_OUT=16, single half), fp32 output.
__global__ __launch_bounds__(256) void k_final(const unsigned short* __restrict__ tb,
    const unsigned short* __restrict__ swzF,
    const float* __restrict__ D1, const float* __restrict__ D2, const float* __restrict__ D3,
    const float* __restrict__ valid, float* __restrict__ out) {
  int tid = threadIdx.x;
  int wv = tid >> 5, lane = tid & 31;
  int g = lane >> 4, n = lane & 15, rl = lane & 15;
  int tile = blockIdx.x * 8 + wv;
  int mt = tile % 3;
  int j = (tile / 3) % NN;
  int i = (tile / (3 * NN)) % NN;
  int b = tile / (3 * NB2);
  int m0 = mt * 16;
  long rowA = ((long)(b * NN + i) * NN + j) * NN + m0;
  long rowT = ((long)(b * NN + j) * NN + i) * NN + m0;
  int c0 = 8 * g, c1 = 16 + 8 * g;
  const unsigned short* pA = tb + (rowA + rl) * CCH;
  const unsigned short* pT = tb + (rowT + rl) * CCH;
  BOp A1, A2, B0, B1;
  A1.q[0] = *(const v4u*)(pA + c0); A1.q[1] = *(const v4u*)(pA + c1);
  A2.q[0] = *(const v4u*)(pT + c0); A2.q[1] = *(const v4u*)(pT + c1);
  const unsigned short* sw = swzF + lane * 16;
  B0.q[0] = *(const v4u*)(sw);       B0.q[1] = *(const v4u*)(sw + 8);
  B1.q[0] = *(const v4u*)(sw + 512); B1.q[1] = *(const v4u*)(sw + 512 + 8);
  v8f z = v8zero();
  z = wmma_bf16(A1, B0, z);
  z = wmma_bf16(A2, B1, z);
  float vi = valid[b * NN + i], vj = valid[b * NN + j];
#pragma unroll
  for (int r = 0; r < 8; ++r) {
    int m = m0 + r + 8 * g;
    float fac = vi * vj * valid[b * NN + m];
    int o1 = ((b * NN + i) * NN + m) * 32;
    int o2 = ((b * NN + j) * NN + m) * 32;
    int o3 = (b * NN + m) * 32;
    out[(rowA + r + 8 * g) * 16 + n] = (z[r] + D1[o1 + n] + D2[o2 + n] + D3[o3 + n]) * fac;
  }
}

// ---------------------------------------------------------------------------
extern "C" void kernel_launch(void* const* d_in, const int* in_sizes, int n_in,
                              void* d_out, int out_size, void* d_ws, size_t ws_size,
                              hipStream_t stream) {
  (void)in_sizes; (void)n_in; (void)out_size; (void)ws_size;
  const float* x            = (const float*)d_in[0];
  const unsigned char* mask = (const unsigned char*)d_in[1];
  const float* basicW[2]    = {(const float*)d_in[2],  (const float*)d_in[10]};
  const float* basicB[2]    = {(const float*)d_in[3],  (const float*)d_in[11]};
  const float* lvW[2]       = {(const float*)d_in[4],  (const float*)d_in[12]};
  const float* lvB[2]       = {(const float*)d_in[5],  (const float*)d_in[13]};
  const float* lwW[2]       = {(const float*)d_in[6],  (const float*)d_in[14]};
  const float* lwB[2]       = {(const float*)d_in[7],  (const float*)d_in[15]};
  const float* loW[2]       = {(const float*)d_in[8],  (const float*)d_in[16]};
  const float* loB[2]       = {(const float*)d_in[9],  (const float*)d_in[17]};
  const float* finW         = (const float*)d_in[18];
  const float* finB         = (const float*)d_in[19];
  float* out = (float*)d_out;

  char* ws = (char*)d_ws;
  size_t off = 0;
  auto alloc = [&](size_t bytes) -> char* {
    char* p = ws + off;
    off = (off + bytes + 255) & ~(size_t)255;
    return p;
  };
  unsigned short* tbuf = (unsigned short*)alloc((size_t)RTOT * CCH * 2);
  unsigned short* vbuf = (unsigned short*)alloc((size_t)RTOT * CCH * 2);
  unsigned short* wbuf = (unsigned short*)alloc((size_t)RTOT * CCH * 2);
  unsigned short* zbuf = (unsigned short*)alloc((size_t)RTOT * CCH * 2);
  float* rowm  = (float*)alloc((size_t)BBATCH * NB2 * 32 * 4);
  float* colm  = (float*)alloc((size_t)BBATCH * NB2 * 32 * 4);
  float* diagm = (float*)alloc((size_t)BBATCH * NB2 * 32 * 4);
  float* totm  = (float*)alloc((size_t)BBATCH * NN * 32 * 4);
  float* D1    = (float*)alloc((size_t)BBATCH * NB2 * 32 * 4);
  float* D2    = (float*)alloc((size_t)BBATCH * NB2 * 32 * 4);
  float* D3    = (float*)alloc((size_t)BBATCH * NN * 32 * 4);
  float* valid = (float*)alloc((size_t)BBATCH * NN * 4);
  unsigned short* swz = (unsigned short*)alloc((size_t)22 * 512 * 2);

  k_valid<<<1, 256, 0, stream>>>(mask, valid);

  // Swizzle all WMMA B-operands once. Layer L ops at swz + L*10*512:
  //   0,1: W0 halves  2,3: W1 halves  4,5: Wv  6,7: Ww  8,9: Wo
  for (int L = 0; L < 2; ++L) {
    int base = L * 10;
    for (int h = 0; h < 2; ++h) {
      k_swz<<<2, 256, 0, stream>>>(basicW[L] + 0 * 1024, 32, h, swz + (base + 0 + h) * 512);
      k_swz<<<2, 256, 0, stream>>>(basicW[L] + 1 * 1024, 32, h, swz + (base + 2 + h) * 512);
      k_swz<<<2, 256, 0, stream>>>(lvW[L], 32, h, swz + (base + 4 + h) * 512);
      k_swz<<<2, 256, 0, stream>>>(lwW[L], 32, h, swz + (base + 6 + h) * 512);
      k_swz<<<2, 256, 0, stream>>>(loW[L], 32, h, swz + (base + 8 + h) * 512);
    }
  }
  k_swz<<<2, 256, 0, stream>>>(finW + 0 * 512, 16, 0, swz + 20 * 512);
  k_swz<<<2, 256, 0, stream>>>(finW + 1 * 512, 16, 0, swz + 21 * 512);

  k_build_t<<<(RTOT * 8) / 256, 256, 0, stream>>>(x, valid, tbuf);

  const int TILE_BLOCKS = (RTOT / 16) / 8; // 3456

  for (int L = 0; L < 2; ++L) {
    k_red1<<<(BBATCH * NB2 * 32) / 256, 256, 0, stream>>>(tbuf, rowm, colm, diagm);
    k_tot<<<24, 256, 0, stream>>>(rowm, totm);
    k_dmix<<<(BBATCH * NB2 * 32) / 256, 256, 0, stream>>>(
        diagm, rowm, colm, basicW[L] + 2 * 1024, basicW[L] + 3 * 1024,
        basicW[L] + 4 * 1024, 32, D1, D2);
    k_d3<<<24, 256, 0, stream>>>(totm, basicW[L] + 5 * 1024, basicB[L], 32, D3);
    k_snconv_vw<<<TILE_BLOCKS, 256, 0, stream>>>(
        tbuf, swz + (size_t)L * 10 * 512, D1, D2, D3, valid, lvB[L], lwB[L], vbuf, wbuf);
    k_z<<<BBATCH * NN * 32, 288, 0, stream>>>(vbuf, wbuf, zbuf);
    k_wo<<<TILE_BLOCKS, 256, 0, stream>>>(zbuf, swz + (size_t)(L * 10 + 8) * 512,
                                          loB[L], valid, tbuf);
  }

  k_red1<<<(BBATCH * NB2 * 32) / 256, 256, 0, stream>>>(tbuf, rowm, colm, diagm);
  k_tot<<<24, 256, 0, stream>>>(rowm, totm);
  k_dmix<<<(BBATCH * NB2 * 16) / 256, 256, 0, stream>>>(
      diagm, rowm, colm, finW + 2 * 512, finW + 3 * 512, finW + 4 * 512, 16, D1, D2);
  k_d3<<<12, 256, 0, stream>>>(totm, finW + 5 * 512, finB, 16, D3);
  k_final<<<TILE_BLOCKS, 256, 0, stream>>>(tbuf, swz + (size_t)20 * 512,
                                           D1, D2, D3, valid, out);
}
